// DopDense_56573309223137
// MI455X (gfx1250) — matrix operand order repaired and checked by reading, not compiled, not run on hardware
//
#include <hip/hip_runtime.h>

typedef __attribute__((ext_vector_type(2)))  float  v2f;
typedef __attribute__((ext_vector_type(8)))  float  v8f;
typedef __attribute__((ext_vector_type(16))) __bf16 v16bf;
typedef __attribute__((ext_vector_type(8)))  __bf16 v8bf;

#define BATCH  4096
#define N_IN   4096
#define UNITS  4096
#define N_DOP  64

// ---------------------------------------------------------------------------
// bf16 split helpers (round-to-nearest-even)
// ---------------------------------------------------------------------------
__device__ __forceinline__ unsigned short bf16_rne(float x) {
    unsigned u = __float_as_uint(x);
    u += 0x7FFFu + ((u >> 16) & 1u);
    return (unsigned short)(u >> 16);
}
__device__ __forceinline__ float bf16_f32(unsigned short h) {
    return __uint_as_float(((unsigned)h) << 16);
}

// ---------------------------------------------------------------------------
// Kernel 1: per-dop-column factor  f[j] = (delta>0) ? 1 + 10*delta : 1
// ---------------------------------------------------------------------------
__global__ __launch_bounds__(256) void dop_factors_kernel(
    const float* __restrict__ w, const float* __restrict__ w_old,
    const int* __restrict__ dop, float* __restrict__ f)
{
    const int j   = blockIdx.x;
    const int col = dop[j];
    float s = 0.0f;
    for (int i = threadIdx.x; i < N_IN; i += 256)
        s += fabsf(w[(size_t)i * UNITS + col] - w_old[(size_t)i * UNITS + col]);
    __shared__ float red[256];
    red[threadIdx.x] = s;
    __syncthreads();
    for (int st = 128; st > 0; st >>= 1) {
        if (threadIdx.x < st) red[threadIdx.x] += red[threadIdx.x + st];
        __syncthreads();
    }
    if (threadIdx.x == 0) {
        float delta = red[0] / (float)N_IN;
        f[j] = (delta > 0.0f) ? (1.0f + 10.0f * delta) : 1.0f;
    }
}

// ---------------------------------------------------------------------------
// Kernel 2: mult[c] = prod_j factor contributions at neighbor columns
// ---------------------------------------------------------------------------
__global__ __launch_bounds__(256) void build_mult_kernel(
    const int* __restrict__ dop, const float* __restrict__ f,
    float* __restrict__ mult)
{
    __shared__ int   sd[N_DOP];
    __shared__ float sf[N_DOP];
    if (threadIdx.x < N_DOP) {
        sd[threadIdx.x] = dop[threadIdx.x];
        sf[threadIdx.x] = f[threadIdx.x];
    }
    __syncthreads();
    const int c = blockIdx.x * 256 + threadIdx.x;
    if (c >= UNITS) return;
    float m = 1.0f;
    for (int j = 0; j < N_DOP; ++j) {
        const int dj = sd[j];
        bool lfree = true, rfree = true;
        for (int k = 0; k < N_DOP; ++k) {
            if (sd[k] == dj - 1) lfree = false;
            if (sd[k] == dj + 1) rfree = false;
        }
        const int lcol = (dj - 1 + UNITS) & (UNITS - 1);
        const int rcol = (dj + 1) & (UNITS - 1);
        if (lfree && lcol == c) m *= sf[j];
        if (rfree && rcol == c) m *= sf[j];
    }
    mult[c] = m;
}

// ---------------------------------------------------------------------------
// Kernel P1: split X (f32) -> Xhi, Xlo (bf16), same [m][k] layout
// ---------------------------------------------------------------------------
__global__ __launch_bounds__(256) void split_x_kernel(
    const float* __restrict__ X,
    unsigned short* __restrict__ Xhi, unsigned short* __restrict__ Xlo)
{
    const size_t i4 = (size_t)blockIdx.x * 256 + threadIdx.x;  // float4 index
    const float4 v = ((const float4*)X)[i4];
    ushort4 h, l;
    h.x = bf16_rne(v.x); l.x = bf16_rne(v.x - bf16_f32(h.x));
    h.y = bf16_rne(v.y); l.y = bf16_rne(v.y - bf16_f32(h.y));
    h.z = bf16_rne(v.z); l.z = bf16_rne(v.z - bf16_f32(h.z));
    h.w = bf16_rne(v.w); l.w = bf16_rne(v.w - bf16_f32(h.w));
    ((ushort4*)Xhi)[i4] = h;
    ((ushort4*)Xlo)[i4] = l;
}

// ---------------------------------------------------------------------------
// Kernel P2: Ws = w * mult; split -> WThi, WTlo stored TRANSPOSED [n][k]
//   64x64 tiles, LDS transpose, coalesced in and out.
// ---------------------------------------------------------------------------
#define TP 68
__global__ __launch_bounds__(256) void split_w_kernel(
    const float* __restrict__ W, const float* __restrict__ mult,
    unsigned short* __restrict__ WThi, unsigned short* __restrict__ WTlo)
{
    __shared__ unsigned short shi[64][TP];
    __shared__ unsigned short slo[64][TP];
    const int tid = threadIdx.x;
    const int k0 = blockIdx.y * 64;
    const int n0 = blockIdx.x * 64;
#pragma unroll
    for (int r = 0; r < 4; ++r) {
        const int idx = tid + r * 256;       // 0..1023
        const int kr  = idx >> 4;            // 0..63
        const int n4  = (idx & 15) * 4;      // 0..60
        const float4 v = *(const float4*)(W + (size_t)(k0 + kr) * UNITS + n0 + n4);
        const float s0 = v.x * mult[n0 + n4 + 0];
        const float s1 = v.y * mult[n0 + n4 + 1];
        const float s2 = v.z * mult[n0 + n4 + 2];
        const float s3 = v.w * mult[n0 + n4 + 3];
        unsigned short h;
        h = bf16_rne(s0); shi[n4 + 0][kr] = h; slo[n4 + 0][kr] = bf16_rne(s0 - bf16_f32(h));
        h = bf16_rne(s1); shi[n4 + 1][kr] = h; slo[n4 + 1][kr] = bf16_rne(s1 - bf16_f32(h));
        h = bf16_rne(s2); shi[n4 + 2][kr] = h; slo[n4 + 2][kr] = bf16_rne(s2 - bf16_f32(h));
        h = bf16_rne(s3); shi[n4 + 3][kr] = h; slo[n4 + 3][kr] = bf16_rne(s3 - bf16_f32(h));
    }
    __syncthreads();
#pragma unroll
    for (int r = 0; r < 4; ++r) {
        const int idx = tid + r * 256;
        const int n   = idx >> 4;            // 0..63
        const int k4  = (idx & 15) * 4;      // 0..60
        *(ushort4*)(WThi + (size_t)(n0 + n) * N_IN + k0 + k4) = *(const ushort4*)&shi[n][k4];
        *(ushort4*)(WTlo + (size_t)(n0 + n) * N_IN + k0 + k4) = *(const ushort4*)&slo[n][k4];
    }
}

// ---------------------------------------------------------------------------
// Kernel 3a (fast path): out = relu(Xsplit @ Wsplit^T + b)
//   via V_WMMA_F32_16X16X32_BF16, 3-pass split accumulation.
//   block 128x128, BK=64, 8 waves (wave 64x32), LDS double-buffered,
//   register-staged global->LDS pipeline.
// ---------------------------------------------------------------------------
#define BM 128
#define BN 128
#define BKB 64
#define PK 72     // LDS pitch (bf16): 144B rows -> 16B aligned, conflict-free b128

union Frag16 { v16bf f; v8bf h[2]; };

__global__ __launch_bounds__(256) void gemm_bf16x3_kernel(
    const unsigned short* __restrict__ Xhi, const unsigned short* __restrict__ Xlo,
    const unsigned short* __restrict__ WThi, const unsigned short* __restrict__ WTlo,
    const float* __restrict__ bias, float* __restrict__ out)
{
    __shared__ unsigned short sA[2][2][BM][PK];   // [buf][hi/lo][m][k]
    __shared__ unsigned short sB[2][2][BN][PK];   // [buf][hi/lo][n][k]
    __shared__ float s_bias[BN];

    const int tid   = threadIdx.x;
    const int lane  = tid & 31;
    const int wave  = tid >> 5;
    const int wm    = wave >> 2;     // 0..1
    const int wn    = wave & 3;      // 0..3
    const int lhalf = lane >> 4;
    const int lrow  = lane & 15;

    const int mbase = blockIdx.y * BM;
    const int nbase = blockIdx.x * BN;

    if (tid < BN) s_bias[tid] = bias[nbase + tid];

    // staging registers: 4 chunks per array
    uint4 stAh[4], stAl[4], stBh[4], stBl[4];

    auto loadTiles = [&](int k0) {
#pragma unroll
        for (int r = 0; r < 4; ++r) {
            const int idx = tid + r * 256;       // 0..1023 16B-chunks
            const int row = idx >> 3;            // 0..127
            const int c   = (idx & 7) * 8;       // 0..56 (bf16 units)
            stAh[r] = *(const uint4*)(Xhi  + (size_t)(mbase + row) * N_IN + k0 + c);
            stAl[r] = *(const uint4*)(Xlo  + (size_t)(mbase + row) * N_IN + k0 + c);
            stBh[r] = *(const uint4*)(WThi + (size_t)(nbase + row) * N_IN + k0 + c);
            stBl[r] = *(const uint4*)(WTlo + (size_t)(nbase + row) * N_IN + k0 + c);
        }
    };
    auto storeTiles = [&](int buf) {
#pragma unroll
        for (int r = 0; r < 4; ++r) {
            const int idx = tid + r * 256;
            const int row = idx >> 3;
            const int c   = (idx & 7) * 8;
            *(uint4*)&sA[buf][0][row][c] = stAh[r];
            *(uint4*)&sA[buf][1][row][c] = stAl[r];
            *(uint4*)&sB[buf][0][row][c] = stBh[r];
            *(uint4*)&sB[buf][1][row][c] = stBl[r];
        }
    };

    v8f acc[4][2] = {};

    loadTiles(0);
    storeTiles(0);
    loadTiles(BKB);
    __syncthreads();

    const int nIter = N_IN / BKB;    // 64
    for (int i = 0; i < nIter; ++i) {
        const int buf = i & 1;
        // ---- compute: 2 K-chunks of 32 -> 2 * 24 = 48 bf16 WMMAs
#pragma unroll
        for (int kc = 0; kc < BKB; kc += 32) {
            Frag16 bh[2], bl[2];
#pragma unroll
            for (int ni = 0; ni < 2; ++ni) {
                const int col = wn * 32 + ni * 16 + lrow;
                const int kb  = kc + lhalf * 8;       // B: K 0-7/16-23 low, 8-15/24-31 high
                bh[ni].h[0] = *(const v8bf*)&sB[buf][0][col][kb];
                bh[ni].h[1] = *(const v8bf*)&sB[buf][0][col][kb + 16];
                bl[ni].h[0] = *(const v8bf*)&sB[buf][1][col][kb];
                bl[ni].h[1] = *(const v8bf*)&sB[buf][1][col][kb + 16];
            }
#pragma unroll
            for (int mi = 0; mi < 4; ++mi) {
                const int row = wm * 64 + mi * 16 + lrow;
                const int ka  = kc + lhalf * 8;       // A: K 0-7/16-23 low, 8-15/24-31 high
                Frag16 ah, al;
                ah.h[0] = *(const v8bf*)&sA[buf][0][row][ka];
                ah.h[1] = *(const v8bf*)&sA[buf][0][row][ka + 16];
                al.h[0] = *(const v8bf*)&sA[buf][1][row][ka];
                al.h[1] = *(const v8bf*)&sA[buf][1][row][ka + 16];
#pragma unroll
                for (int ni = 0; ni < 2; ++ni) {
                    acc[mi][ni] = __builtin_amdgcn_wmma_f32_16x16x32_bf16(
                        false, ah.f, false, bh[ni].f, (short)0, acc[mi][ni], false, false);
                    acc[mi][ni] = __builtin_amdgcn_wmma_f32_16x16x32_bf16(
                        false, ah.f, false, bl[ni].f, (short)0, acc[mi][ni], false, false);
                    acc[mi][ni] = __builtin_amdgcn_wmma_f32_16x16x32_bf16(
                        false, al.f, false, bh[ni].f, (short)0, acc[mi][ni], false, false);
                }
            }
        }
        // ---- pipeline: commit staged tile i+1, start loading tile i+2
        if (i + 1 < nIter) {
            __syncthreads();
            storeTiles(buf ^ 1);
            if (i + 2 < nIter) loadTiles((i + 2) * BKB);
            __syncthreads();
        }
    }

    // ---- epilogue: bias + relu
#pragma unroll
    for (int mi = 0; mi < 4; ++mi) {
#pragma unroll
        for (int ni = 0; ni < 2; ++ni) {
            const int tile_m = mbase + wm * 64 + mi * 16;
            const int nloc   = wn * 32 + ni * 16 + lrow;
            const int gcol   = nbase + nloc;
            const float bb   = s_bias[nloc];
#pragma unroll
            for (int r = 0; r < 8; ++r) {
                const int grow = tile_m + r + lhalf * 8;
                float v = acc[mi][ni][r] + bb;
                out[(size_t)grow * UNITS + gcol] = v > 0.0f ? v : 0.0f;
            }
        }
    }
}

// ---------------------------------------------------------------------------
// Kernel 3b (fallback, small ws): fp32 WMMA GEMM (single-buffered)
// ---------------------------------------------------------------------------
#define BK 32
#define LDP 36
__global__ __launch_bounds__(256) void gemm_wmma_f32_kernel(
    const float* __restrict__ X, const float* __restrict__ W,
    const float* __restrict__ bias, const float* __restrict__ mult,
    float* __restrict__ out)
{
    __shared__ float sA[BM][LDP];
    __shared__ float sB[BN][LDP];
    __shared__ float s_mult[BN];
    __shared__ float s_bias[BN];

    const int tid   = threadIdx.x;
    const int lane  = tid & 31;
    const int wave  = tid >> 5;
    const int wm    = wave >> 2;
    const int wn    = wave & 3;
    const int lhalf = lane >> 4;
    const int lrow  = lane & 15;
    const int mbase = blockIdx.y * BM;
    const int nbase = blockIdx.x * BN;

    if (tid < BN) { s_mult[tid] = mult[nbase + tid]; s_bias[tid] = bias[nbase + tid]; }
    __syncthreads();

    v8f acc[4][2] = {};
    for (int k0 = 0; k0 < N_IN; k0 += BK) {
#pragma unroll
        for (int r = 0; r < 4; ++r) {
            const int idx = tid + r * 256;
            const int row = idx >> 3;
            const int c4  = (idx & 7) * 4;
            *(float4*)&sA[row][c4] =
                *(const float4*)(X + (size_t)(mbase + row) * N_IN + k0 + c4);
        }
#pragma unroll
        for (int r = 0; r < 4; ++r) {
            const int idx = tid + r * 256;
            const int kr  = idx >> 5;
            const int n4  = (idx & 31) * 4;
            const float4 v = *(const float4*)(W + (size_t)(k0 + kr) * UNITS + nbase + n4);
            sB[n4 + 0][kr] = v.x * s_mult[n4 + 0];
            sB[n4 + 1][kr] = v.y * s_mult[n4 + 1];
            sB[n4 + 2][kr] = v.z * s_mult[n4 + 2];
            sB[n4 + 3][kr] = v.w * s_mult[n4 + 3];
        }
        __syncthreads();
#pragma unroll
        for (int ks = 0; ks < BK; ks += 4) {
            const int kk = ks + lhalf * 2;
            v2f afrag[4], bfrag[2];
#pragma unroll
            for (int mi = 0; mi < 4; ++mi)
                afrag[mi] = *(const v2f*)&sA[wm * 64 + mi * 16 + lrow][kk];
#pragma unroll
            for (int ni = 0; ni < 2; ++ni)
                bfrag[ni] = *(const v2f*)&sB[wn * 32 + ni * 16 + lrow][kk];
#pragma unroll
            for (int mi = 0; mi < 4; ++mi)
#pragma unroll
                for (int ni = 0; ni < 2; ++ni)
                    acc[mi][ni] = __builtin_amdgcn_wmma_f32_16x16x4_f32(
                        false, afrag[mi], false, bfrag[ni], (short)0, acc[mi][ni], false, false);
        }
        __syncthreads();
    }
#pragma unroll
    for (int mi = 0; mi < 4; ++mi)
#pragma unroll
        for (int ni = 0; ni < 2; ++ni) {
            const int tile_m = mbase + wm * 64 + mi * 16;
            const int nloc   = wn * 32 + ni * 16 + lrow;
            const float bb   = s_bias[nloc];
#pragma unroll
            for (int r = 0; r < 8; ++r) {
                float v = acc[mi][ni][r] + bb;
                out[(size_t)(tile_m + r + lhalf * 8) * UNITS + nbase + nloc] =
                    v > 0.0f ? v : 0.0f;
            }
        }
}

// ---------------------------------------------------------------------------
extern "C" void kernel_launch(void* const* d_in, const int* in_sizes, int n_in,
                              void* d_out, int out_size, void* d_ws, size_t ws_size,
                              hipStream_t stream) {
    const float* x     = (const float*)d_in[0];
    const float* w     = (const float*)d_in[1];
    const float* bias  = (const float*)d_in[2];
    const float* w_old = (const float*)d_in[3];
    const int*   dop   = (const int*)d_in[4];

    float* f    = (float*)d_ws;
    float* mult = f + 64;

    dop_factors_kernel<<<N_DOP, 256, 0, stream>>>(w, w_old, dop, f);
    build_mult_kernel<<<UNITS / 256, 256, 0, stream>>>(dop, f, mult);

    const size_t arr  = (size_t)BATCH * N_IN * sizeof(unsigned short);  // 32MB
    const size_t need = 4096 + 4 * arr;

    if (ws_size >= need) {
        unsigned short* Xhi  = (unsigned short*)((char*)d_ws + 4096);
        unsigned short* Xlo  = (unsigned short*)((char*)d_ws + 4096 + arr);
        unsigned short* WThi = (unsigned short*)((char*)d_ws + 4096 + 2 * arr);
        unsigned short* WTlo = (unsigned short*)((char*)d_ws + 4096 + 3 * arr);

        split_x_kernel<<<(BATCH * N_IN / 4) / 256, 256, 0, stream>>>(x, Xhi, Xlo);
        split_w_kernel<<<dim3(UNITS / 64, N_IN / 64), 256, 0, stream>>>(w, mult, WThi, WTlo);

        dim3 grid(UNITS / BN, BATCH / BM);
        gemm_bf16x3_kernel<<<grid, 256, 0, stream>>>(Xhi, Xlo, WThi, WTlo, bias,
                                                     (float*)d_out);
    } else {
        dim3 grid(UNITS / BN, BATCH / BM);
        gemm_wmma_f32_kernel<<<grid, 256, 0, stream>>>(x, w, bias, mult, (float*)d_out);
    }
}
